// Token_performer_38551626449208
// MI455X (gfx1250) — compile-verified, matrix operand order checked by reading, not run on hardware
//
#include <hip/hip_runtime.h>
#include <hip/hip_bf16.h>
#include <math.h>

typedef __attribute__((ext_vector_type(16))) __bf16          v16bf;
typedef __attribute__((ext_vector_type(16))) unsigned short  v16u;
typedef __attribute__((ext_vector_type(8)))  float           v8f;

union FragU {
  v16u  v;
  v16bf bf;
  unsigned int u[8];
};

__device__ __forceinline__ unsigned short f2bf(float f) {
  unsigned int u = __float_as_uint(f);
  u += 0x7fffu + ((u >> 16) & 1u);   // round-to-nearest-even
  return (unsigned short)(u >> 16);
}
__device__ __forceinline__ float bf2f(unsigned short h) {
  return __uint_as_float(((unsigned int)h) << 16);
}

// ---- CDNA5 async global->LDS staging (ASYNCcnt-tracked, no VGPR round-trip)
#if defined(__has_builtin)
#  if __has_builtin(__builtin_amdgcn_global_load_async_to_lds_b128)
#    define HAVE_ASYNC_LDS 1
#  endif
#endif

#ifdef HAVE_ASYNC_LDS
typedef int v4i __attribute__((ext_vector_type(4)));
typedef __attribute__((address_space(1))) v4i* as1_v4i;
typedef __attribute__((address_space(3))) v4i* as3_v4i;
#endif

__device__ __forceinline__ void async_ld128(const unsigned short* g,
                                            unsigned short* l) {
#ifdef HAVE_ASYNC_LDS
  __builtin_amdgcn_global_load_async_to_lds_b128(
      (as1_v4i)(v4i*)(void*)const_cast<unsigned short*>(g),
      (as3_v4i)(v4i*)(void*)l, 0, 0);
#else
  *(uint4*)l = *(const uint4*)g;
#endif
}
__device__ __forceinline__ void wait_async() {
#ifdef HAVE_ASYNC_LDS
#  if __has_builtin(__builtin_amdgcn_s_wait_asynccnt)
  __builtin_amdgcn_s_wait_asynccnt(0);
#  else
  asm volatile("s_wait_asynccnt 0" ::: "memory");
#  endif
#endif
}

// ---------------------------------------------------------------------------
// Generic NT GEMM:  C[M,N] = A[M,K](bf16) * W[N,K]^T(bf16)  (+ epilogue)
// BM=128, BN=64, BK=32; 256 threads = 8 waves, each wave a 32x32 patch.
// Double-buffered LDS, async tensor staging.
// ---------------------------------------------------------------------------
#define BM 128
#define BN 64
#define BKK 32
#define AS_LD 40
#define WS_LD 40

__global__ __launch_bounds__(256) void gemm_nt_bf16(
    const unsigned short* __restrict__ A, int lda,
    const unsigned short* __restrict__ W, int ldw,
    long long wBatchStride, int rowsPerBatch,
    const float* __restrict__ bias,
    int op, const float* __restrict__ rowvec, float opc,
    const float* __restrict__ residF, const unsigned short* __restrict__ residB,
    int ldr,
    float* __restrict__ outF, unsigned short* __restrict__ outB,
    int N, int K)
{
  __shared__ __align__(16) unsigned short As[2 * BM * AS_LD];
  __shared__ __align__(16) unsigned short Ws[2 * BN * WS_LD];

  const int tid   = threadIdx.x;
  const int lane  = tid & 31;
  const int wave  = tid >> 5;
  const int waveM = wave >> 1;   // 0..3
  const int waveN = wave & 1;    // 0..1
  const int rowBase = blockIdx.y * BM;
  const int colBase = blockIdx.x * BN;

  if (wBatchStride) W += (long long)(rowBase / rowsPerBatch) * wBatchStride;

  const int half = lane >> 4;
  const int rl   = lane & 15;
  int kb[8];
#pragma unroll
  for (int i = 0; i < 8; ++i)
    kb[i] = ((i < 4) ? 2 * i : 8 + 2 * i) + 8 * half;

  // per-thread staging coordinates (fixed)
  const int ar0 = tid >> 2,          ao0 = (tid & 3) << 3;
  const int ar1 = (tid + 256) >> 2,  ao1 = ((tid + 256) & 3) << 3;
  const int wr  = tid >> 2,          wo  = (tid & 3) << 3;

  auto stage = [&](int buf, int k0) {
    unsigned short* Ab = &As[buf * BM * AS_LD];
    unsigned short* Wb = &Ws[buf * BN * WS_LD];
    async_ld128(&A[(size_t)(rowBase + ar0) * lda + (k0 + ao0)], &Ab[ar0 * AS_LD + ao0]);
    async_ld128(&A[(size_t)(rowBase + ar1) * lda + (k0 + ao1)], &Ab[ar1 * AS_LD + ao1]);
    async_ld128(&W[(size_t)(colBase + wr) * ldw + (k0 + wo)],   &Wb[wr * WS_LD + wo]);
  };

  const v8f vzero = {0.f, 0.f, 0.f, 0.f, 0.f, 0.f, 0.f, 0.f};
  v8f acc[2][2];
  acc[0][0] = vzero; acc[0][1] = vzero; acc[1][0] = vzero; acc[1][1] = vzero;

  stage(0, 0);
  int buf = 0;
  for (int k0 = 0; k0 < K; k0 += BKK) {
    wait_async();          // this wave's staged tile has landed in LDS
    __syncthreads();       // every wave's tile visible
    if (k0 + BKK < K) {
      stage(buf ^ 1, k0 + BKK);                       // fill other buffer
      __builtin_prefetch(&A[(size_t)(rowBase + ar0) * lda + (k0 + 2 * BKK)], 0, 1);
    }

    const unsigned short* Ab = &As[buf * BM * AS_LD];
    const unsigned short* Wb = &Ws[buf * BN * WS_LD];
    FragU af[2], wf[2];
#pragma unroll
    for (int tm = 0; tm < 2; ++tm) {
      const unsigned short* p = &Ab[(waveM * 32 + tm * 16 + rl) * AS_LD];
#pragma unroll
      for (int i = 0; i < 8; ++i)
        af[tm].u[i] = *(const unsigned int*)(p + kb[i]);
    }
#pragma unroll
    for (int tn = 0; tn < 2; ++tn) {
      const unsigned short* p = &Wb[(waveN * 32 + tn * 16 + rl) * WS_LD];
#pragma unroll
      for (int i = 0; i < 8; ++i)
        wf[tn].u[i] = *(const unsigned int*)(p + kb[i]);
    }
#pragma unroll
    for (int tm = 0; tm < 2; ++tm)
#pragma unroll
      for (int tn = 0; tn < 2; ++tn)
        acc[tm][tn] = __builtin_amdgcn_wmma_f32_16x16x32_bf16(
            false, af[tm].bf, false, wf[tn].bf,
            (short)0, acc[tm][tn], false, false);
    __syncthreads();       // all reads of `buf` done before it is restaged
    buf ^= 1;
  }

  // epilogue
#pragma unroll
  for (int tm = 0; tm < 2; ++tm) {
#pragma unroll
    for (int tn = 0; tn < 2; ++tn) {
      int gr0 = rowBase + waveM * 32 + tm * 16 + 8 * half;
      int gc  = colBase + waveN * 32 + tn * 16 + rl;
      float bv = bias ? bias[gc] : 0.f;
#pragma unroll
      for (int i = 0; i < 8; ++i) {
        int gr  = gr0 + i;
        float v = acc[tm][tn][i] + bv;
        if (op == 1)       v = __expf(v - rowvec[gr]) * opc;          // FAVOR+
        else if (op == 2)  v = 0.5f * v * (1.f + erff(v * 0.70710678118654752f));
        else if (op == 3)  v = v / (rowvec[gr] + opc);                // / (D+eps)
        if (residF)        v += residF[(size_t)gr * ldr + gc];
        else if (residB)   v += bf2f(residB[(size_t)gr * ldr + gc]);
        size_t o = (size_t)gr * N + gc;
        if (outF) outF[o] = v;
        if (outB) outB[o] = f2bf(v);
      }
    }
  }
}

// ---------------------------------------------------------------------------
// TN GEMM (per-batch):  C[M,N] = sum_k A[k,M] * B[k,N]   (both K-major bf16)
// Used for kptv = v^T @ kp  (M=768, N=384, K=4096 per batch).
// ---------------------------------------------------------------------------
#define AS2_LD 136
#define BS2_LD 72

__global__ __launch_bounds__(256) void gemm_tn_bf16(
    const unsigned short* __restrict__ A, int lda, long long aBatch,
    const unsigned short* __restrict__ Bm, int ldb, long long bBatch,
    float* __restrict__ outF, unsigned short* __restrict__ outB, long long oBatch,
    int N, int K)
{
  __shared__ __align__(16) unsigned short As[2 * BKK * AS2_LD];
  __shared__ __align__(16) unsigned short Bs[2 * BKK * BS2_LD];

  const int tid   = threadIdx.x;
  const int lane  = tid & 31;
  const int wave  = tid >> 5;
  const int waveM = wave >> 1;
  const int waveN = wave & 1;
  const int rowBase = blockIdx.y * BM;   // M (hd) base
  const int colBase = blockIdx.x * BN;   // N (m) base

  A    += (long long)blockIdx.z * aBatch;
  Bm   += (long long)blockIdx.z * bBatch;
  outF += (long long)blockIdx.z * oBatch;
  outB += (long long)blockIdx.z * oBatch;

  const int half = lane >> 4;
  const int rl   = lane & 15;
  int kb[8];
#pragma unroll
  for (int i = 0; i < 8; ++i)
    kb[i] = ((i < 4) ? 2 * i : 8 + 2 * i) + 8 * half;

  const int ar0 = tid >> 4,          ao0 = (tid & 15) << 3;
  const int ar1 = (tid + 256) >> 4,  ao1 = ((tid + 256) & 15) << 3;
  const int br  = tid >> 3,          bo  = (tid & 7) << 3;

  auto stage = [&](int buf, int k0) {
    unsigned short* Ab = &As[buf * BKK * AS2_LD];
    unsigned short* Bb = &Bs[buf * BKK * BS2_LD];
    async_ld128(&A[(size_t)(k0 + ar0) * lda + (rowBase + ao0)], &Ab[ar0 * AS2_LD + ao0]);
    async_ld128(&A[(size_t)(k0 + ar1) * lda + (rowBase + ao1)], &Ab[ar1 * AS2_LD + ao1]);
    async_ld128(&Bm[(size_t)(k0 + br) * ldb + (colBase + bo)],  &Bb[br * BS2_LD + bo]);
  };

  const v8f vzero = {0.f, 0.f, 0.f, 0.f, 0.f, 0.f, 0.f, 0.f};
  v8f acc[2][2];
  acc[0][0] = vzero; acc[0][1] = vzero; acc[1][0] = vzero; acc[1][1] = vzero;

  stage(0, 0);
  int buf = 0;
  for (int k0 = 0; k0 < K; k0 += BKK) {
    wait_async();
    __syncthreads();
    if (k0 + BKK < K) stage(buf ^ 1, k0 + BKK);

    const unsigned short* Ab = &As[buf * BKK * AS2_LD];
    const unsigned short* Bb = &Bs[buf * BKK * BS2_LD];
    FragU af[2], bfr[2];
#pragma unroll
    for (int tm = 0; tm < 2; ++tm) {
      int mIdx = waveM * 32 + tm * 16 + rl;
#pragma unroll
      for (int i = 0; i < 8; ++i) {
        unsigned int s0 = Ab[kb[i] * AS2_LD + mIdx];
        unsigned int s1 = Ab[(kb[i] + 1) * AS2_LD + mIdx];
        af[tm].u[i] = s0 | (s1 << 16);
      }
    }
#pragma unroll
    for (int tn = 0; tn < 2; ++tn) {
      int nIdx = waveN * 32 + tn * 16 + rl;
#pragma unroll
      for (int i = 0; i < 8; ++i) {
        unsigned int s0 = Bb[kb[i] * BS2_LD + nIdx];
        unsigned int s1 = Bb[(kb[i] + 1) * BS2_LD + nIdx];
        bfr[tn].u[i] = s0 | (s1 << 16);
      }
    }
#pragma unroll
    for (int tm = 0; tm < 2; ++tm)
#pragma unroll
      for (int tn = 0; tn < 2; ++tn)
        acc[tm][tn] = __builtin_amdgcn_wmma_f32_16x16x32_bf16(
            false, af[tm].bf, false, bfr[tn].bf,
            (short)0, acc[tm][tn], false, false);
    __syncthreads();
    buf ^= 1;
  }

#pragma unroll
  for (int tm = 0; tm < 2; ++tm) {
#pragma unroll
    for (int tn = 0; tn < 2; ++tn) {
      int gr0 = rowBase + waveM * 32 + tm * 16 + 8 * half;
      int gc  = colBase + waveN * 32 + tn * 16 + rl;
#pragma unroll
      for (int i = 0; i < 8; ++i) {
        int gr = gr0 + i;
        size_t o = (size_t)gr * N + gc;
        float v = acc[tm][tn][i];
        outF[o] = v;
        outB[o] = f2bf(v);
      }
    }
  }
}

// ---------------------------------------------------------------------------
// Small kernels
// ---------------------------------------------------------------------------
__global__ void cast_f32_bf16(const float* __restrict__ s,
                              unsigned short* __restrict__ d, int n) {
  int i = blockIdx.x * blockDim.x + threadIdx.x;
  if (i < n) d[i] = f2bf(s[i]);
}

// LayerNorm over C, output bf16. One block (256 thr) per row.
__global__ __launch_bounds__(256) void ln_cast_kernel(
    const float* __restrict__ x, const float* __restrict__ g,
    const float* __restrict__ b, unsigned short* __restrict__ out, int C)
{
  const int row = blockIdx.x;
  const float* xr = x + (size_t)row * C;
  float s = 0.f, s2 = 0.f;
  for (int c = threadIdx.x; c < C; c += 256) { float v = xr[c]; s += v; s2 += v * v; }
  __shared__ float red[16];
  int lane = threadIdx.x & 31, wave = threadIdx.x >> 5;
  for (int o = 16; o; o >>= 1) { s += __shfl_down(s, o, 32); s2 += __shfl_down(s2, o, 32); }
  if (lane == 0) { red[wave] = s; red[8 + wave] = s2; }
  __syncthreads();
  if (threadIdx.x == 0) {
    float S = 0.f, S2 = 0.f;
    for (int w = 0; w < 8; ++w) { S += red[w]; S2 += red[8 + w]; }
    red[0] = S; red[1] = S2;
  }
  __syncthreads();
  float mean = red[0] / C;
  float var  = red[1] / C - mean * mean;
  float rstd = rsqrtf(var + 1e-5f);
  unsigned short* orow = out + (size_t)row * C;
  for (int c = threadIdx.x; c < C; c += 256)
    orow[c] = f2bf((xr[c] - mean) * rstd * g[c] + b[c]);
}

// xd_{k,q}[t] = 0.5 * sum_i z^2 over the 768-wide k and q slices of kqv(bf16).
__global__ __launch_bounds__(256) void xd_kernel(
    const unsigned short* __restrict__ kqv, float* __restrict__ xdk,
    float* __restrict__ xdq)
{
  const int row = blockIdx.x;
  const unsigned short* r = kqv + (size_t)row * 2304;
  float sk = 0.f, sq = 0.f;
  for (int c = threadIdx.x; c < 768; c += 256) {
    float kv = bf2f(r[c]);        sk += kv * kv;
    float qv = bf2f(r[768 + c]);  sq += qv * qv;
  }
  __shared__ float red[16];
  int lane = threadIdx.x & 31, wave = threadIdx.x >> 5;
  for (int o = 16; o; o >>= 1) { sk += __shfl_down(sk, o, 32); sq += __shfl_down(sq, o, 32); }
  if (lane == 0) { red[wave] = sk; red[8 + wave] = sq; }
  __syncthreads();
  if (threadIdx.x == 0) {
    float Sk = 0.f, Sq = 0.f;
    for (int w = 0; w < 8; ++w) { Sk += red[w]; Sq += red[8 + w]; }
    xdk[row] = 0.5f * Sk;
    xdq[row] = 0.5f * Sq;
  }
}

// ksum[b][m] = sum_t kp[b,t,m]
__global__ void ksum_kernel(const float* __restrict__ kp, float* __restrict__ ksum) {
  int i = blockIdx.x * blockDim.x + threadIdx.x;   // 8*384
  if (i >= 8 * 384) return;
  int b = i / 384, m = i % 384;
  const float* p = kp + ((size_t)b * 4096) * 384 + m;
  float s = 0.f;
  for (int t = 0; t < 4096; ++t) s += p[(size_t)t * 384];
  ksum[i] = s;
}

// D[t] = qp[t,:] . ksum[b,:]   (one wave per row)
__global__ __launch_bounds__(256) void d_kernel(
    const float* __restrict__ qp, const float* __restrict__ ks,
    float* __restrict__ D)
{
  int wave = threadIdx.x >> 5, lane = threadIdx.x & 31;
  int row = blockIdx.x * 8 + wave;
  const float* qr = qp + (size_t)row * 384;
  const float* kb = ks + (size_t)(row / 4096) * 384;
  float s = 0.f;
  for (int m = lane; m < 384; m += 32) s += qr[m] * kb[m];
  for (int o = 16; o; o >>= 1) s += __shfl_down(s, o, 32);
  if (lane == 0) D[row] = s;
}

// ---------------------------------------------------------------------------
// Host: orchestrate the pipeline
// ---------------------------------------------------------------------------
extern "C" void kernel_launch(void* const* d_in, const int* in_sizes, int n_in,
                              void* d_out, int out_size, void* d_ws, size_t ws_size,
                              hipStream_t stream) {
  (void)in_sizes; (void)n_in; (void)out_size; (void)ws_size;
  const float* x      = (const float*)d_in[0];
  const float* kqv_w  = (const float*)d_in[1];
  const float* kqv_b  = (const float*)d_in[2];
  const float* proj_w = (const float*)d_in[3];
  const float* proj_b = (const float*)d_in[4];
  const float* n1_w   = (const float*)d_in[5];
  const float* n1_b   = (const float*)d_in[6];
  const float* n2_w   = (const float*)d_in[7];
  const float* n2_b   = (const float*)d_in[8];
  const float* m1_w   = (const float*)d_in[9];
  const float* m1_b   = (const float*)d_in[10];
  const float* m2_w   = (const float*)d_in[11];
  const float* m2_b   = (const float*)d_in[12];
  const float* wfav   = (const float*)d_in[13];
  float* out = (float*)d_out;

  const int T = 32768;            // B*N tokens
  const int Dd = 768, Mm = 384, KQV = 2304;
  const float INV_SQRT_M = 0.05103103630798288f;  // 1/sqrt(384)

  char* ws = (char*)d_ws;
  size_t off = 0;
  auto alloc = [&](size_t bytes) -> char* {
    off = (off + 255) & ~(size_t)255;
    char* p = ws + off; off += bytes; return p;
  };

  unsigned short* x1b    = (unsigned short*)alloc((size_t)T * Dd * 2);
  unsigned short* kqvb   = (unsigned short*)alloc((size_t)T * KQV * 2);
  unsigned short* kqvwb  = (unsigned short*)alloc((size_t)KQV * Dd * 2);
  unsigned short* wfavb  = (unsigned short*)alloc((size_t)Mm * Dd * 2);
  unsigned short* projwb = (unsigned short*)alloc((size_t)Dd * Dd * 2);
  unsigned short* m1wb   = (unsigned short*)alloc((size_t)Dd * Dd * 2);
  unsigned short* m2wb   = (unsigned short*)alloc((size_t)Dd * Dd * 2);
  float*          xdk    = (float*)alloc((size_t)T * 4);
  float*          xdq    = (float*)alloc((size_t)T * 4);
  float*          kpf    = (float*)alloc((size_t)T * Mm * 4);
  unsigned short* kpb    = (unsigned short*)alloc((size_t)T * Mm * 2);
  float*          qpf    = (float*)alloc((size_t)T * Mm * 4);
  unsigned short* qpb    = (unsigned short*)alloc((size_t)T * Mm * 2);
  float*          ksum   = (float*)alloc((size_t)8 * Mm * 4);
  float*          Dv     = (float*)alloc((size_t)T * 4);
  float*          kptvf  = (float*)alloc((size_t)8 * Dd * Mm * 4);
  unsigned short* kptvb  = (unsigned short*)alloc((size_t)8 * Dd * Mm * 2);
  unsigned short* yattb  = (unsigned short*)alloc((size_t)T * Dd * 2);
  unsigned short* x2b    = (unsigned short*)alloc((size_t)T * Dd * 2);
  unsigned short* hdnb   = (unsigned short*)alloc((size_t)T * Dd * 2);

  // 1) weight casts f32 -> bf16
  auto cast = [&](const float* s, unsigned short* d, int n) {
    cast_f32_bf16<<<(n + 255) / 256, 256, 0, stream>>>(s, d, n);
  };
  cast(kqv_w, kqvwb, KQV * Dd);
  cast(wfav,  wfavb, Mm * Dd);
  cast(proj_w, projwb, Dd * Dd);
  cast(m1_w,  m1wb, Dd * Dd);
  cast(m2_w,  m2wb, Dd * Dd);

  // 2) LN1 -> x1 (bf16)
  ln_cast_kernel<<<T, 256, 0, stream>>>(x, n1_w, n1_b, x1b, Dd);

  // 3) kqv = x1 @ kqv_w^T + kqv_b  -> bf16 [T,2304]
  gemm_nt_bf16<<<dim3(KQV / BN, T / BM), 256, 0, stream>>>(
      x1b, Dd, kqvwb, Dd, 0, 4096, kqv_b,
      0, nullptr, 0.f, nullptr, nullptr, 0,
      nullptr, kqvb, KQV, Dd);

  // 4) xd_k, xd_q
  xd_kernel<<<T, 256, 0, stream>>>(kqvb, xdk, xdq);

  // 5) qp = exp(q @ w^T - xd_q)/sqrt(m)   (f32 + bf16)
  gemm_nt_bf16<<<dim3(Mm / BN, T / BM), 256, 0, stream>>>(
      kqvb + 768, KQV, wfavb, Dd, 0, 4096, nullptr,
      1, xdq, INV_SQRT_M, nullptr, nullptr, 0,
      qpf, qpb, Mm, Dd);
  // 6) kp likewise
  gemm_nt_bf16<<<dim3(Mm / BN, T / BM), 256, 0, stream>>>(
      kqvb + 0, KQV, wfavb, Dd, 0, 4096, nullptr,
      1, xdk, INV_SQRT_M, nullptr, nullptr, 0,
      kpf, kpb, Mm, Dd);

  // 7) ksum over tokens per batch
  ksum_kernel<<<(8 * Mm + 255) / 256, 256, 0, stream>>>(kpf, ksum);
  // 8) D = qp . ksum
  d_kernel<<<T / 8, 256, 0, stream>>>(qpf, ksum, Dv);

  // 9) kptv[b] = v[b]^T @ kp[b]   (TN, per batch; v lives at col 1536 of kqv)
  gemm_tn_bf16<<<dim3(Mm / BN, Dd / BM, 8), 256, 0, stream>>>(
      kqvb + 1536, KQV, (long long)4096 * KQV,
      kpb, Mm, (long long)4096 * Mm,
      kptvf, kptvb, (long long)Dd * Mm,
      Mm, 4096);

  // 10) y_att = (qp @ kptv^T) / (D + eps)  -> bf16  (per-batch weight)
  gemm_nt_bf16<<<dim3(Dd / BN, T / BM), 256, 0, stream>>>(
      qpb, Mm, kptvb, Mm, (long long)Dd * Mm, 4096, nullptr,
      3, Dv, 1e-8f, nullptr, nullptr, 0,
      nullptr, yattb, Dd, Mm);

  // 11) y = v + y_att @ proj_w^T + proj_b  -> f32 in d_out
  gemm_nt_bf16<<<dim3(Dd / BN, T / BM), 256, 0, stream>>>(
      yattb, Dd, projwb, Dd, 0, 4096, proj_b,
      0, nullptr, 0.f, nullptr, kqvb + 1536, KQV,
      out, nullptr, Dd, Dd);

  // 12) x2 = LN2(y) -> bf16
  ln_cast_kernel<<<T, 256, 0, stream>>>(out, n2_w, n2_b, x2b, Dd);

  // 13) hdn = gelu(x2 @ m1^T + m1_b) -> bf16
  gemm_nt_bf16<<<dim3(Dd / BN, T / BM), 256, 0, stream>>>(
      x2b, Dd, m1wb, Dd, 0, 4096, m1_b,
      2, nullptr, 0.f, nullptr, nullptr, 0,
      nullptr, hdnb, Dd, Dd);

  // 14) out = y + hdn @ m2^T + m2_b   (in-place residual on d_out)
  gemm_nt_bf16<<<dim3(Dd / BN, T / BM), 256, 0, stream>>>(
      hdnb, Dd, m2wb, Dd, 0, 4096, m2_b,
      0, nullptr, 0.f, out, nullptr, Dd,
      out, nullptr, Dd, Dd);
}
